// CatalyticDiffusionModel_17188459119292
// MI455X (gfx1250) — matrix operand order repaired and checked by reading, not compiled, not run on hardware
//
#include <hip/hip_runtime.h>
#include <cstdint>

#define USE_ASYNC_LDS 1

typedef __attribute__((ext_vector_type(16))) _Float16 v16h;
typedef __attribute__((ext_vector_type(8)))  _Float16 v8h;
typedef __attribute__((ext_vector_type(8)))  float    v8f;

#define HDIM    256
#define NNODES  8192
#define NEDGES  262144
#define NLAYERS 6
#define TSTEPS  1000

__device__ __forceinline__ float siluf(float x){ return x / (1.f + __expf(-x)); }
__device__ __forceinline__ float sigmf(float x){ return 1.f / (1.f + __expf(-x)); }

// D = A(16x32 f16) * B(32x16 f16) + C(f32)   -- wave32 WMMA
__device__ __forceinline__ v8f wmma_step(v16h a, v16h b, v8f c){
  return __builtin_amdgcn_wmma_f32_16x16x32_f16(false, a, false, b, (short)0, c, false, false);
}

// A-fragment from a row-major f16 row in LDS.  Per ISA layout for 16-bit A 16x32:
// lane L (M = L%16, hi = L/16): halves 0..7 -> K = kt*32 + 8*hi + h,
//                               halves 8..15 -> K = kt*32 + 16 + 8*hi + (h-8)
__device__ __forceinline__ v16h load_a_frag(const _Float16* row, int kt, int hi){
  const v8h* p0 = (const v8h*)(row + kt*32 + 8*hi);
  const v8h* p1 = (const v8h*)(row + kt*32 + 16 + 8*hi);
  v8h lo = *p0, hh = *p1;
  return __builtin_shufflevector(lo, hh, 0,1,2,3,4,5,6,7,8,9,10,11,12,13,14,15);
}

// B-fragment from pre-packed weights: frag = 32 lanes x 16 consecutive halves.
__device__ __forceinline__ v16h load_b_frag(const _Float16* pack, int frag, int lane){
  return *(const v16h*)(pack + (size_t)frag*512 + lane*16);
}

// CDNA5 async global->LDS 16B copy (ASYNCcnt-tracked), fallback = plain 16B copy.
__device__ __forceinline__ void lds_cp16(void* lds, const void* g){
#if USE_ASYNC_LDS
  unsigned l32 = (unsigned)(uintptr_t)lds;                 // flat LDS addr: low 32 bits are the LDS offset
  unsigned long long g64 = (unsigned long long)(uintptr_t)g;
  asm volatile("global_load_async_to_lds_b128 %0, %1, off" :: "v"(l32), "v"(g64) : "memory");
#else
  *(uint4*)lds = *(const uint4*)g;
#endif
}
__device__ __forceinline__ void lds_cp_wait(){
#if USE_ASYNC_LDS
  asm volatile("s_wait_asynccnt 0" ::: "memory");
#endif
}

// ---------------------------------------------------------------------------
// Repack W[K x 256] (f32, row-major, x@W convention) into WMMA-B fragment order.
// frag f = n_tile*(K/32) + k_tile ; within frag: lane (N=lane%16, hi=lane/16)
// x 16 halves (h<8: K = kt*32 + 8*hi + h ; h>=8: K = kt*32 + 16 + 8*hi + (h-8)).
// ---------------------------------------------------------------------------
__global__ void k_pack(const float* __restrict__ wsrc, _Float16* __restrict__ dst, int K){
  int id = blockIdx.x*256 + threadIdx.x;       // total = K*256
  int f    = id >> 9;
  int r    = id & 511;
  int lane = r >> 4, h = r & 15;
  int ktiles = K >> 5;
  int kt = f % ktiles, nt = f / ktiles;
  int k = kt*32 + ((h < 8) ? (8*(lane >> 4) + h) : (16 + 8*(lane >> 4) + (h - 8)));
  int n = nt*16 + (lane & 15);
  dst[id] = (_Float16)wsrc[k*256 + n];
}

// ---------------------------------------------------------------------------
// Precompute (1 block, 256 threads): diffusion schedule, time + condition MLPs.
// ---------------------------------------------------------------------------
struct PreArgs {
  const int* t; const float* anchor; const float* distc; const float* coordc;
  const float* t1b; const float* t1w; const float* t2b; const float* t2w;
  const float* tpb; const float* tpw;
  const float* a1b; const float* a1w; const float* a2b; const float* a2w;
  const float* d1b; const float* d1w; const float* d2b; const float* d2w;
  const float* c1b; const float* c1w; const float* c2b; const float* c2w;
  const float* f1b; const float* f1w; const float* flg; const float* flb;
  const float* f2b; const float* f2w; const float* cpb; const float* cpw;
  float* add_out; float* sched_out;
};
__global__ __launch_bounds__(256) void k_pre(PreArgs A){
  __shared__ float te[256], h1[256], temb[256], tproj[256];
  __shared__ float anc[256], dstv[256], crdv[256], mh[256];
  __shared__ float f1[256], stats[2];
  int c = threadIdx.x;
  if (c == 0){
    int tt = *A.t; float ac = 1.f;
    for (int i=0;i<=tt;i++){
      float beta = 1e-4f + (0.02f-1e-4f)*((float)i/(float)(TSTEPS-1));
      ac *= (1.f-beta);
    }
    A.sched_out[0] = sqrtf(ac); A.sched_out[1] = sqrtf(1.f-ac);
  }
  float tf = (float)(*A.t);
  const float scl = logf(10000.f)/127.f;
  te[c] = (c<128) ? sinf(tf*__expf(-scl*(float)c)) : cosf(tf*__expf(-scl*(float)(c-128)));
  __syncthreads();
  { float s=A.t1b[c]; for (int k=0;k<256;k++) s += te[k]*A.t1w[k*256+c]; h1[c]=siluf(s); } __syncthreads();
  { float s=A.t2b[c]; for (int k=0;k<256;k++) s += h1[k]*A.t2w[k*256+c]; temb[c]=s; }     __syncthreads();
  { float s=A.tpb[c]; for (int k=0;k<256;k++) s += temb[k]*A.tpw[k*256+c]; tproj[c]=s; }  __syncthreads();
  // anchor: mean over 8 rows of silu(lin1) ; mean commutes through the affine lin2
  { float m=0.f;
    for (int r=0;r<8;r++){ float s=A.a1b[c]; for (int j=0;j<16;j++) s += A.anchor[r*16+j]*A.a1w[j*256+c]; m += siluf(s); }
    mh[c]=m*0.125f; } __syncthreads();
  { float s=A.a2b[c]; for (int k=0;k<256;k++) s += mh[k]*A.a2w[k*256+c]; anc[c]=s; } __syncthreads();
  if (c<128){ float m=0.f;
    for (int r=0;r<16;r++){ float s=A.d1b[c]; for (int j=0;j<4;j++) s += A.distc[r*4+j]*A.d1w[j*128+c]; m += siluf(s); }
    mh[c]=m*(1.f/16.f); } __syncthreads();
  { float s=A.d2b[c]; for (int k=0;k<128;k++) s += mh[k]*A.d2w[k*256+c]; dstv[c]=s; } __syncthreads();
  if (c<128){ float m=0.f;
    for (int r=0;r<8;r++){ float s=A.c1b[c]; for (int j=0;j<3;j++) s += A.coordc[r*3+j]*A.c1w[j*128+c]; m += siluf(s); }
    mh[c]=m*0.125f; } __syncthreads();
  { float s=A.c2b[c]; for (int k=0;k<128;k++) s += mh[k]*A.c2w[k*256+c]; crdv[c]=s; } __syncthreads();
  { float s=A.f1b[c];
    for (int k=0;k<256;k++)
      s += anc[k]*A.f1w[k*256+c] + dstv[k]*A.f1w[(256+k)*256+c] + crdv[k]*A.f1w[(512+k)*256+c];
    f1[c]=s; } __syncthreads();
  if (c==0){
    float mu=0.f; for (int k=0;k<256;k++) mu += f1[k]; mu *= (1.f/256.f);
    float v=0.f;  for (int k=0;k<256;k++){ float d=f1[k]-mu; v += d*d; } v *= (1.f/256.f);
    stats[0]=mu; stats[1]=rsqrtf(v+1e-5f);
  } __syncthreads();
  { float y=(f1[c]-stats[0])*stats[1]*A.flg[c]+A.flb[c]; h1[c]=siluf(y); } __syncthreads();
  { float s=A.f2b[c]; for (int k=0;k<256;k++) s += h1[k]*A.f2w[k*256+c]; temb[c]=s; } __syncthreads();
  { float s=A.cpb[c]; for (int k=0;k<256;k++) s += temb[k]*A.cpw[k*256+c]; A.add_out[c]=tproj[c]+s; }
}

// ---------------------------------------------------------------------------
// Node init: h = embed[atom] + (t_proj+c_proj), f16 mirror, q_sample coords,
// zero the per-layer accumulators.
// ---------------------------------------------------------------------------
__global__ __launch_bounds__(256) void k_init(const int* __restrict__ at, const float* __restrict__ coords,
    const float* __restrict__ noise, const float* __restrict__ emb, const float* __restrict__ addv,
    const float* __restrict__ sched, float* __restrict__ h32, _Float16* __restrict__ hq,
    float* __restrict__ x, float* __restrict__ m_i, float* __restrict__ xacc){
  int n = blockIdx.x, c = threadIdx.x;
  size_t id = (size_t)n*256 + c;
  float v = emb[at[n]*256 + c] + addv[c];
  h32[id] = v; hq[id] = (_Float16)v; m_i[id] = 0.f;
  if (c < 3){
    x[n*3+c] = sched[0]*coords[n*3+c] + sched[1]*noise[n*3+c];
    xacc[n*3+c] = 0.f;
  }
}

// ---------------------------------------------------------------------------
// Edge kernel: 16 edges/block, 8 waves, full edge-MLP chain on WMMA.
// ---------------------------------------------------------------------------
struct EdgeArgs {
  const int* row; const int* col;
  const _Float16* hq; const float* x;
  const _Float16* e1p; const float* e1b; const float* e1w512;
  const _Float16* e2p; const float* e2b;
  const _Float16* c1p; const float* c1b; const float* c2w;
  const float* attw; const float* attb;
  float* m_i; float* xacc;
};
__global__ __launch_bounds__(256) void k_edge(EdgeArgs A){
  __shared__ alignas(32) _Float16 Ain[16][512];   // [h_row | h_col] f16
  __shared__ alignas(32) _Float16 M1[16][256];
  __shared__ alignas(32) _Float16 M2[16][256];
  __shared__ float M2f[16][256];
  __shared__ float spart[16][16];
  __shared__ float sdist[16], srel[16][3], satt[16], scw[16];
  __shared__ int   srow[16], scol[16];

  const int t = threadIdx.x;
  const int eb = blockIdx.x * 16;
  if (t < 16){
    int e = eb + t;
    int r = A.row[e], c = A.col[e];
    srow[t]=r; scol[t]=c;
    float dx = A.x[r*3+0]-A.x[c*3+0];
    float dy = A.x[r*3+1]-A.x[c*3+1];
    float dz = A.x[r*3+2]-A.x[c*3+2];
    srel[t][0]=dx; srel[t][1]=dy; srel[t][2]=dz;
    sdist[t]=sqrtf(dx*dx+dy*dy+dz*dz);
    scw[t]=0.f;
  }
  __syncthreads();
  // async stage 16KB of gathered h into LDS (1024 x 16B chunks)
  #pragma unroll
  for (int i=0;i<4;i++){
    int cidx = t + i*256;
    int e    = cidx >> 6;
    int off8 = (cidx & 63) * 8;                  // half offset within 512
    int node = (off8 < 256) ? srow[e] : scol[e];
    lds_cp16(&Ain[e][off8], A.hq + (size_t)node*256 + (off8 & 255));
  }
  lds_cp_wait();
  __syncthreads();

  const int l = t & 31, w = t >> 5;
  const int hi = l >> 4, nc = l & 15, n0 = w*32;

  // GEMM1: e_in(16x512) @ W1(512x256), + dist * W1row512 + bias, silu
  v8f acc0 = {}, acc1 = {};
  for (int kt=0; kt<16; ++kt){
    v16h a = load_a_frag(&Ain[nc][0], kt, hi);
    acc0 = wmma_step(a, load_b_frag(A.e1p, (w*2+0)*16 + kt, l), acc0);
    acc1 = wmma_step(a, load_b_frag(A.e1p, (w*2+1)*16 + kt, l), acc1);
  }
  {
    float w5a=A.e1w512[n0+nc],   b1a=A.e1b[n0+nc];
    float w5b=A.e1w512[n0+16+nc],b1b=A.e1b[n0+16+nc];
    #pragma unroll
    for (int v=0; v<8; ++v){
      int M = v + 8*hi;
      M1[M][n0+nc]    = (_Float16)siluf(acc0[v] + sdist[M]*w5a + b1a);
      M1[M][n0+16+nc] = (_Float16)siluf(acc1[v] + sdist[M]*w5b + b1b);
    }
  }
  __syncthreads();
  // GEMM2: m(16x256) @ W2(256x256) + b, silu  -> M2 (f16) + M2f (f32)
  acc0 = (v8f){}; acc1 = (v8f){};
  for (int kt=0; kt<8; ++kt){
    v16h a = load_a_frag(&M1[nc][0], kt, hi);
    acc0 = wmma_step(a, load_b_frag(A.e2p, (w*2+0)*8 + kt, l), acc0);
    acc1 = wmma_step(a, load_b_frag(A.e2p, (w*2+1)*8 + kt, l), acc1);
  }
  {
    float ba=A.e2b[n0+nc], bb=A.e2b[n0+16+nc];
    #pragma unroll
    for (int v=0; v<8; ++v){
      int M = v + 8*hi;
      float z0=siluf(acc0[v]+ba), z1=siluf(acc1[v]+bb);
      M2[M][n0+nc]=(_Float16)z0;    M2f[M][n0+nc]=z0;
      M2[M][n0+16+nc]=(_Float16)z1; M2f[M][n0+16+nc]=z1;
    }
  }
  __syncthreads();
  // attention partial dots (16 edges x 16 partials)
  {
    int e = t & 15, part = t >> 4;
    float p = 0.f;
    for (int j=0;j<16;++j){ int c = part*16 + j; p += M2f[e][c]*A.attw[c]; }
    spart[e][part] = p;
  }
  // GEMM3: coord1(m) -> silu -> dot coord2  (per-edge scalar via LDS atomics)
  acc0 = (v8f){}; acc1 = (v8f){};
  for (int kt=0; kt<8; ++kt){
    v16h a = load_a_frag(&M2[nc][0], kt, hi);
    acc0 = wmma_step(a, load_b_frag(A.c1p, (w*2+0)*8 + kt, l), acc0);
    acc1 = wmma_step(a, load_b_frag(A.c1p, (w*2+1)*8 + kt, l), acc1);
  }
  {
    float ba=A.c1b[n0+nc], bb=A.c1b[n0+16+nc];
    float wa=A.c2w[n0+nc],  wb=A.c2w[n0+16+nc];
    #pragma unroll
    for (int v=0; v<8; ++v){
      int M = v + 8*hi;
      atomicAdd(&scw[M], siluf(acc0[v]+ba)*wa + siluf(acc1[v]+bb)*wb);
    }
  }
  __syncthreads();
  if (t < 16){
    float s = A.attb[0];
    for (int p=0;p<16;++p) s += spart[t][p];
    satt[t] = sigmf(s);
  }
  __syncthreads();
  // scatter att*m into m_i (global f32 atomics -> global_atomic_add_f32)
  {
    int e = t >> 4, c0 = t & 15;
    float ae = satt[e];
    size_t rb = (size_t)srow[e]*256;
    #pragma unroll
    for (int j=0;j<16;++j){
      int c = c0 + j*16;
      unsafeAtomicAdd(&A.m_i[rb + c], ae * M2f[e][c]);
    }
  }
  if (t < 48){
    int e = t/3, d = t - 3*e;
    float cd = srel[e][d] / (sdist[e] + 1e-8f);
    unsafeAtomicAdd(&A.xacc[(size_t)srow[e]*3 + d], scw[e]*cd);
  }
}

// ---------------------------------------------------------------------------
// Node kernel: 16 nodes/block; [h|m_i] -> node1 -> silu -> node2 -> residual.
// Also applies coord deltas and re-zeros accumulators for the next layer.
// ---------------------------------------------------------------------------
struct NodeArgs {
  const _Float16* n1p; const float* n1b;
  const _Float16* n2p; const float* n2b;
  float* h32; _Float16* hq; float* m_i; float* x; float* xacc;
};
__global__ __launch_bounds__(256) void k_node(NodeArgs A){
  __shared__ alignas(32) _Float16 Ain[16][512];
  __shared__ alignas(32) _Float16 H1[16][256];
  const int t = threadIdx.x;
  const int nb = blockIdx.x*16;
  #pragma unroll
  for (int i=0;i<2;++i){
    int cidx = t + i*256;
    int nd = cidx >> 5;
    int off8 = (cidx & 31) * 8;
    lds_cp16(&Ain[nd][off8], A.hq + (size_t)(nb+nd)*256 + off8);
  }
  lds_cp_wait();
  #pragma unroll
  for (int j=0;j<16;++j){
    int idx = t + j*256;
    int nd = idx >> 8, c = idx & 255;
    Ain[nd][256+c] = (_Float16)A.m_i[(size_t)(nb+nd)*256 + c];
  }
  __syncthreads();
  const int l=t&31, w=t>>5, hi=l>>4, nc=l&15, n0=w*32;
  v8f acc0 = {}, acc1 = {};
  for (int kt=0;kt<16;++kt){
    v16h a = load_a_frag(&Ain[nc][0], kt, hi);
    acc0 = wmma_step(a, load_b_frag(A.n1p, (w*2+0)*16 + kt, l), acc0);
    acc1 = wmma_step(a, load_b_frag(A.n1p, (w*2+1)*16 + kt, l), acc1);
  }
  {
    float ba=A.n1b[n0+nc], bb=A.n1b[n0+16+nc];
    #pragma unroll
    for (int v=0; v<8; ++v){
      int M = v + 8*hi;
      H1[M][n0+nc]    = (_Float16)siluf(acc0[v]+ba);
      H1[M][n0+16+nc] = (_Float16)siluf(acc1[v]+bb);
    }
  }
  __syncthreads();
  acc0 = (v8f){}; acc1 = (v8f){};
  for (int kt=0;kt<8;++kt){
    v16h a = load_a_frag(&H1[nc][0], kt, hi);
    acc0 = wmma_step(a, load_b_frag(A.n2p, (w*2+0)*8 + kt, l), acc0);
    acc1 = wmma_step(a, load_b_frag(A.n2p, (w*2+1)*8 + kt, l), acc1);
  }
  {
    float ba=A.n2b[n0+nc], bb=A.n2b[n0+16+nc];
    #pragma unroll
    for (int v=0; v<8; ++v){
      int M = v + 8*hi;
      size_t g0 = (size_t)(nb+M)*256 + n0 + nc;
      size_t g1 = g0 + 16;
      float hv0 = A.h32[g0] + acc0[v] + ba;
      float hv1 = A.h32[g1] + acc1[v] + bb;
      A.h32[g0]=hv0; A.hq[g0]=(_Float16)hv0;
      A.h32[g1]=hv1; A.hq[g1]=(_Float16)hv1;
    }
  }
  if (t < 48){
    int nd=t/3, d=t-3*nd;
    size_t xi=(size_t)(nb+nd)*3+d;
    A.x[xi] += A.xacc[xi];
    A.xacc[xi] = 0.f;
  }
  #pragma unroll
  for (int j=0;j<16;++j){
    int idx = t + j*256;
    A.m_i[(size_t)(nb+(idx>>8))*256 + (idx&255)] = 0.f;
  }
}

// ---------------------------------------------------------------------------
// LayerNorm + heads (wave32 shuffle reductions; one wave per node).
// Output: [pred_noise (N*3)] then [pred_atom_logits (N*32)].
// ---------------------------------------------------------------------------
__global__ __launch_bounds__(256) void k_head(const float* __restrict__ h32,
    const float* __restrict__ g, const float* __restrict__ b,
    const float* __restrict__ chw, const float* __restrict__ chb,
    const float* __restrict__ ahw, const float* __restrict__ ahb,
    float* __restrict__ out){
  __shared__ float Y[8][256];
  int t=threadIdx.x, w=t>>5, l=t&31;
  int n = blockIdx.x*8 + w;
  const float* hr = h32 + (size_t)n*256;
  float s=0.f;
  for (int c=l;c<256;c+=32) s += hr[c];
  for (int o=16;o>0;o>>=1) s += __shfl_xor(s,o,32);
  float mu = s*(1.f/256.f);
  float vs=0.f;
  for (int c=l;c<256;c+=32){ float d=hr[c]-mu; vs += d*d; }
  for (int o=16;o>0;o>>=1) vs += __shfl_xor(vs,o,32);
  float rstd = rsqrtf(vs*(1.f/256.f)+1e-5f);
  for (int c=l;c<256;c+=32) Y[w][c] = (hr[c]-mu)*rstd*g[c] + b[c];
  __syncthreads();
  for (int o=l;o<35;o+=32){
    float acc;
    if (o < 3){
      acc = chb[o];
      for (int c=0;c<256;++c) acc += Y[w][c]*chw[c*3+o];
      out[(size_t)n*3+o] = acc;
    } else {
      int a2 = o-3;
      acc = ahb[a2];
      for (int c=0;c<256;++c) acc += Y[w][c]*ahw[c*32+a2];
      out[(size_t)NNODES*3 + (size_t)n*32 + a2] = acc;
    }
  }
}

// ---------------------------------------------------------------------------
// Host side.
// Param leaves assumed appended after the 8 array inputs, flattened in jax
// tree order (sorted dict keys; 'b' < 'g' < 'w'; layers list in order):
//  +0 anc1.b +1 anc1.w +2 anc2.b +3 anc2.w +4 atom_embed +5 atom_head.b
//  +6 atom_head.w +7 cond_proj.b +8 cond_proj.w +9 coord_head.b +10 coord_head.w
//  +11 crd1.b +12 crd1.w +13 crd2.b +14 crd2.w +15 dst1.b +16 dst1.w
//  +17 dst2.b +18 dst2.w +19 fus1.b +20 fus1.w +21 fus2.b +22 fus2.w
//  +23 fus_ln.b +24 fus_ln.g
//  +25.. layers (13 each): att.b att.w coord1.b coord1.w coord2.w
//                          edge1.b edge1.w edge2.b edge2.w node1.b node1.w node2.b node2.w
//  +103 norm.b +104 norm.g +105 time1.b +106 time1.w +107 time2.b +108 time2.w
//  +109 time_proj.b +110 time_proj.w
// ---------------------------------------------------------------------------
extern "C" void kernel_launch(void* const* d_in, const int* in_sizes, int n_in,
                              void* d_out, int out_size, void* d_ws, size_t ws_size,
                              hipStream_t stream){
  (void)in_sizes; (void)n_in; (void)out_size; (void)ws_size;
  const int*   atom_types = (const int*)  d_in[0];
  const float* coords     = (const float*)d_in[1];
  const int*   edge_index = (const int*)  d_in[2];
  const float* anchorf    = (const float*)d_in[3];
  const float* distc      = (const float*)d_in[4];
  const float* coordc     = (const float*)d_in[5];
  const float* noise      = (const float*)d_in[6];
  const int*   tptr       = (const int*)  d_in[7];
  const int PB = 8;
  auto PF = [&](int i){ return (const float*)d_in[PB + i]; };
  enum {
    ANC1_B=0, ANC1_W, ANC2_B, ANC2_W, ATOM_EMBED, ATOM_HEAD_B, ATOM_HEAD_W,
    COND_PROJ_B, COND_PROJ_W, COORD_HEAD_B, COORD_HEAD_W,
    CRD1_B, CRD1_W, CRD2_B, CRD2_W, DST1_B, DST1_W, DST2_B, DST2_W,
    FUS1_B, FUS1_W, FUS2_B, FUS2_W, FUSLN_B, FUSLN_G, LAYERS0   // = 25
  };
  enum { L_ATT_B=0, L_ATT_W, L_C1_B, L_C1_W, L_C2_W, L_E1_B, L_E1_W,
         L_E2_B, L_E2_W, L_N1_B, L_N1_W, L_N2_B, L_N2_W };
  const int NORM_B  = LAYERS0 + 13*NLAYERS;   // 103
  const int NORM_G  = NORM_B + 1;
  const int TIME1_B = NORM_G + 1, TIME1_W = TIME1_B+1;
  const int TIME2_B = TIME1_W+1,  TIME2_W = TIME2_B+1;
  const int TIMEP_B = TIME2_W+1,  TIMEP_W = TIMEP_B+1;
  auto LFp = [&](int L, int which){ return (const float*)d_in[PB + LAYERS0 + L*13 + which]; };

  char* base = (char*)d_ws; size_t off = 0;
  auto carve = [&](size_t bytes)->void*{
    void* p = base + off; off = (off + bytes + 255) & ~(size_t)255; return p; };
  float*     addv  = (float*)carve(HDIM*4);
  float*     sched = (float*)carve(64);
  float*     h32   = (float*)carve((size_t)NNODES*HDIM*4);
  _Float16*  hq    = (_Float16*)carve((size_t)NNODES*HDIM*2);
  float*     m_i   = (float*)carve((size_t)NNODES*HDIM*4);
  float*     xbuf  = (float*)carve((size_t)NNODES*3*4);
  float*     xacc  = (float*)carve((size_t)NNODES*3*4);
  _Float16 *e1p[NLAYERS], *e2p[NLAYERS], *c1p[NLAYERS], *n1p[NLAYERS], *n2p[NLAYERS];
  for (int L=0; L<NLAYERS; ++L){
    e1p[L]=(_Float16*)carve((size_t)512*256*2);
    e2p[L]=(_Float16*)carve((size_t)256*256*2);
    c1p[L]=(_Float16*)carve((size_t)256*256*2);
    n1p[L]=(_Float16*)carve((size_t)512*256*2);
    n2p[L]=(_Float16*)carve((size_t)256*256*2);
    k_pack<<<512,256,0,stream>>>(LFp(L,L_E1_W), e1p[L], 512);   // rows 0..511 of 513x256
    k_pack<<<256,256,0,stream>>>(LFp(L,L_E2_W), e2p[L], 256);
    k_pack<<<256,256,0,stream>>>(LFp(L,L_C1_W), c1p[L], 256);
    k_pack<<<512,256,0,stream>>>(LFp(L,L_N1_W), n1p[L], 512);
    k_pack<<<256,256,0,stream>>>(LFp(L,L_N2_W), n2p[L], 256);
  }

  PreArgs P;
  P.t=tptr; P.anchor=anchorf; P.distc=distc; P.coordc=coordc;
  P.t1b=PF(TIME1_B); P.t1w=PF(TIME1_W); P.t2b=PF(TIME2_B); P.t2w=PF(TIME2_W);
  P.tpb=PF(TIMEP_B); P.tpw=PF(TIMEP_W);
  P.a1b=PF(ANC1_B);  P.a1w=PF(ANC1_W);  P.a2b=PF(ANC2_B);  P.a2w=PF(ANC2_W);
  P.d1b=PF(DST1_B);  P.d1w=PF(DST1_W);  P.d2b=PF(DST2_B);  P.d2w=PF(DST2_W);
  P.c1b=PF(CRD1_B);  P.c1w=PF(CRD1_W);  P.c2b=PF(CRD2_B);  P.c2w=PF(CRD2_W);
  P.f1b=PF(FUS1_B);  P.f1w=PF(FUS1_W);  P.flg=PF(FUSLN_G); P.flb=PF(FUSLN_B);
  P.f2b=PF(FUS2_B);  P.f2w=PF(FUS2_W);  P.cpb=PF(COND_PROJ_B); P.cpw=PF(COND_PROJ_W);
  P.add_out=addv; P.sched_out=sched;
  k_pre<<<1,256,0,stream>>>(P);

  k_init<<<NNODES,256,0,stream>>>(atom_types, coords, noise, PF(ATOM_EMBED),
                                  addv, sched, h32, hq, xbuf, m_i, xacc);

  for (int L=0; L<NLAYERS; ++L){
    EdgeArgs EA;
    EA.row = edge_index; EA.col = edge_index + NEDGES;
    EA.hq = hq; EA.x = xbuf;
    EA.e1p = e1p[L]; EA.e1b = LFp(L,L_E1_B); EA.e1w512 = LFp(L,L_E1_W) + (size_t)512*256;
    EA.e2p = e2p[L]; EA.e2b = LFp(L,L_E2_B);
    EA.c1p = c1p[L]; EA.c1b = LFp(L,L_C1_B); EA.c2w = LFp(L,L_C2_W);
    EA.attw = LFp(L,L_ATT_W); EA.attb = LFp(L,L_ATT_B);
    EA.m_i = m_i; EA.xacc = xacc;
    k_edge<<<NEDGES/16,256,0,stream>>>(EA);

    NodeArgs NA;
    NA.n1p = n1p[L]; NA.n1b = LFp(L,L_N1_B);
    NA.n2p = n2p[L]; NA.n2b = LFp(L,L_N2_B);
    NA.h32 = h32; NA.hq = hq; NA.m_i = m_i; NA.x = xbuf; NA.xacc = xacc;
    k_node<<<NNODES/16,256,0,stream>>>(NA);
  }

  k_head<<<NNODES/8,256,0,stream>>>(h32, PF(NORM_G), PF(NORM_B),
                                    PF(COORD_HEAD_W), PF(COORD_HEAD_B),
                                    PF(ATOM_HEAD_W), PF(ATOM_HEAD_B),
                                    (float*)d_out);
}
